// BGNN4VD_24498493456396
// MI455X (gfx1250) — matrix operand order repaired and checked by reading, not compiled, not run on hardware
//
#include <hip/hip_runtime.h>
#include <cstdint>

typedef __attribute__((ext_vector_type(16))) __bf16 v16bf;
typedef __attribute__((ext_vector_type(8)))  float  v8f;

#define NDIM 512
#define LSTR 40        // LDS row stride (elements) = 80B; keeps 16B chunks aligned
#define NEG_SLOPE 0.2f
#define BN_EPS 1e-5f

// ---------------------------------------------------------------- atomics
__device__ __forceinline__ void atomic_max_f32(float* addr, float v) {
  if (v >= 0.0f) atomicMax((int*)addr, __float_as_int(v));
  else           atomicMin((unsigned int*)addr, (unsigned int)__float_as_int(v));
}

// ---------------------------------------------------------------- prep: bf16 convert + weight transpose
__global__ void prep_kernel(const float* __restrict__ x,
                            const float* __restrict__ Wf,
                            const float* __restrict__ Wb,
                            const float* __restrict__ Wfuse,
                            __bf16* __restrict__ xb,
                            __bf16* __restrict__ WcT,
                            __bf16* __restrict__ WfuseT, int N) {
  size_t i = (size_t)blockIdx.x * blockDim.x + threadIdx.x;
  size_t total = (size_t)N * NDIM;
  if (i < total) xb[i] = (__bf16)x[i];
  if (i < (size_t)NDIM * NDIM) {
    int n = (int)(i >> 9);           // output feature
    int k = (int)(i & 511);          // input feature
    float wv = (n < 256) ? Wf[(size_t)k * 256 + n] : Wb[(size_t)k * 256 + (n - 256)];
    WcT[i] = (__bf16)wv;                          // WcT[n][k]
    WfuseT[i] = (__bf16)Wfuse[(size_t)k * NDIM + n];
  }
}

// ---------------------------------------------------------------- workspace init
__global__ void init_ws_kernel(float* __restrict__ Agg, float* __restrict__ Mx,
                               float* __restrict__ Den, float* __restrict__ Bn, int N) {
  size_t i = (size_t)blockIdx.x * blockDim.x + threadIdx.x;
  if (i < (size_t)N * NDIM) Agg[i] = 0.0f;
  if (i < (size_t)N * 8) { Mx[i] = __int_as_float(0xFF800000); Den[i] = 0.0f; }
  if (i < 1024) Bn[i] = 0.0f;
}

// ---------------------------------------------------------------- WMMA GEMM: C[M,512] = A[M,512] * B, BT[n][k] pre-transposed
// Double-buffered LDS, tiles staged with GLOBAL_LOAD_ASYNC_TO_LDS_B128 (ASYNCcnt).
template<bool OUT_BF16>
__global__ __launch_bounds__(256) void wmma_gemm(
    const __bf16* __restrict__ A, const __bf16* __restrict__ BT,
    float* __restrict__ Cf, __bf16* __restrict__ Cb,
    const float* __restrict__ bias, int M) {
  __shared__ __align__(16) __bf16 sA[2][128 * LSTR];
  __shared__ __align__(16) __bf16 sB[2][128 * LSTR];
  const int m0 = blockIdx.x * 128;
  const int n0 = blockIdx.y * 128;
  const int tid  = threadIdx.x;
  const int lane = tid & 31;
  const int wave = tid >> 5;
  const int wm = wave >> 1;   // 0..3 : 32-row band
  const int wn = wave & 1;    // 0..1 : 64-col band
  const int half = lane >> 4;
  const int l15  = lane & 15;

  // issue async copies of one K-slab (A: rows m0.., B: cols n0.., k0..k0+31) into buffer `buf`
  auto issue_tile = [&](int k0, int buf) {
#pragma unroll
    for (int i = 0; i < 2; ++i) {
      int c = tid + i * 256;        // 512 chunks of 16B per tile
      int row = c >> 2;
      int q = c & 3;
      int gr = m0 + row;
      if (gr >= M) gr = 0;          // clamp: padding rows feed only unstored C rows
      const __bf16* ga = A + (size_t)gr * NDIM + k0 + q * 8;
      uint32_t la = (uint32_t)(uintptr_t)(&sA[buf][row * LSTR + q * 8]);
      asm volatile("global_load_async_to_lds_b128 %0, %1, off"
                   :: "v"(la), "v"((uint64_t)(uintptr_t)ga) : "memory");
      const __bf16* gb = BT + (size_t)(n0 + row) * NDIM + k0 + q * 8;
      uint32_t lb = (uint32_t)(uintptr_t)(&sB[buf][row * LSTR + q * 8]);
      asm volatile("global_load_async_to_lds_b128 %0, %1, off"
                   :: "v"(lb), "v"((uint64_t)(uintptr_t)gb) : "memory");
    }
  };

  const v8f vzero = {};
  v8f acc[2][4];
#pragma unroll
  for (int i = 0; i < 2; ++i)
#pragma unroll
    for (int j = 0; j < 4; ++j) acc[i][j] = vzero;

  issue_tile(0, 0);

  for (int k0 = 0; k0 < NDIM; k0 += 32) {
    const int buf = (k0 >> 5) & 1;
    asm volatile("s_wait_asynccnt 0x0" ::: "memory");  // this wave's copies into `buf` done
    __syncthreads();                                    // all waves' copies done / prev reads done
    if (k0 + 32 < NDIM) issue_tile(k0 + 32, buf ^ 1);   // prefetch next slab during compute

    v16bf afr[2], bfr[4];
#pragma unroll
    for (int mt = 0; mt < 2; ++mt) {
      // A 16x32 bf16 frag: lanes 0-15 -> K 0..7 & 16..23 of row l15; lanes 16-31 -> K 8..15 & 24..31
      int roff = (wm * 32 + mt * 16 + l15) * LSTR;
      union { uint4 u[2]; v16bf v; } t;
      t.u[0] = *(const uint4*)(&sA[buf][roff + half * 8]);
      t.u[1] = *(const uint4*)(&sA[buf][roff + 16 + half * 8]);
      afr[mt] = t.v;
    }
#pragma unroll
    for (int nt = 0; nt < 4; ++nt) {
      // B 32x16 bf16 frag: lanes 0-15 -> K 0..7 & 16..23 of col l15; lanes 16-31 -> K 8..15 & 24..31
      int noff = (wn * 64 + nt * 16 + l15) * LSTR;
      union { uint4 u[2]; v16bf v; } t;
      t.u[0] = *(const uint4*)(&sB[buf][noff + half * 8]);
      t.u[1] = *(const uint4*)(&sB[buf][noff + 16 + half * 8]);
      bfr[nt] = t.v;
    }
#pragma unroll
    for (int mt = 0; mt < 2; ++mt)
#pragma unroll
      for (int nt = 0; nt < 4; ++nt)
        acc[mt][nt] = __builtin_amdgcn_wmma_f32_16x16x32_bf16(
            false, afr[mt], false, bfr[nt], (short)0, acc[mt][nt], false, false);
  }

  // epilogue: C/D layout: VGPR r -> M = r + 8*half, N = l15
#pragma unroll
  for (int mt = 0; mt < 2; ++mt) {
#pragma unroll
    for (int nt = 0; nt < 4; ++nt) {
      int col = n0 + wn * 64 + nt * 16 + l15;
      float badd = bias ? bias[col] : 0.0f;
#pragma unroll
      for (int r = 0; r < 8; ++r) {
        int row = m0 + wm * 32 + mt * 16 + half * 8 + r;
        if (row < M) {
          float v = acc[mt][nt][r] + badd;
          if (OUT_BF16) Cb[(size_t)row * NDIM + col] = (__bf16)v;
          else          Cf[(size_t)row * NDIM + col] = v;
        }
      }
    }
  }
}

// ---------------------------------------------------------------- per-node attention scores
// Asc[n][0..3]=a_src_f, [4..7]=a_dst_f, [8..11]=a_src_b, [12..15]=a_dst_b
__global__ void node_scores(const __bf16* __restrict__ H,
                            const float* __restrict__ asf, const float* __restrict__ adf,
                            const float* __restrict__ asb, const float* __restrict__ adb,
                            float* __restrict__ Asc, int N) {
  int gid = blockIdx.x * blockDim.x + threadIdx.x;
  int n = gid >> 2, h = gid & 3;
  if (n >= N) return;
  const __bf16* hf = H + (size_t)n * NDIM + h * 64;
  const __bf16* hb = hf + 256;
  float sf = 0.f, df = 0.f, sb = 0.f, db = 0.f;
#pragma unroll 8
  for (int c = 0; c < 64; ++c) {
    float vf = (float)hf[c];
    float vb = (float)hb[c];
    sf += vf * asf[h * 64 + c];
    df += vf * adf[h * 64 + c];
    sb += vb * asb[h * 64 + c];
    db += vb * adb[h * 64 + c];
  }
  float* o = Asc + (size_t)n * 16;
  o[h] = sf; o[4 + h] = df; o[8 + h] = sb; o[12 + h] = db;
}

// ---------------------------------------------------------------- edge pass 1: leaky-relu scores + segment max
__global__ void edge_pass1(const long long* __restrict__ ei, const float* __restrict__ Asc,
                           float* __restrict__ Ebuf, float* __restrict__ Mx, int E) {
  int e = blockIdx.x * blockDim.x + threadIdx.x;
  if (e >= E) return;
  size_t s = (size_t)ei[e];
  size_t d = (size_t)ei[(size_t)E + e];
  const float* As = Asc + s * 16;
  const float* Ad = Asc + d * 16;
  float* eb = Ebuf + (size_t)e * 8;
#pragma unroll
  for (int h = 0; h < 4; ++h) {
    float t = As[h] + Ad[4 + h];                  // fwd: a_s[src] + a_d[dst], agg at dst
    t = t > 0.f ? t : NEG_SLOPE * t;
    eb[h] = t;
    atomic_max_f32(&Mx[d * 8 + h], t);
    float tb = Ad[8 + h] + As[12 + h];            // bwd: a_s[dst] + a_d[src], agg at src
    tb = tb > 0.f ? tb : NEG_SLOPE * tb;
    eb[4 + h] = tb;
    atomic_max_f32(&Mx[s * 8 + 4 + h], tb);
  }
}

// ---------------------------------------------------------------- edge pass 2: exp + segment sum
__global__ void edge_pass2(const long long* __restrict__ ei, const float* __restrict__ Mx,
                           float* __restrict__ Ebuf, float* __restrict__ Den, int E) {
  int e = blockIdx.x * blockDim.x + threadIdx.x;
  if (e >= E) return;
  size_t s = (size_t)ei[e];
  size_t d = (size_t)ei[(size_t)E + e];
  float* eb = Ebuf + (size_t)e * 8;
#pragma unroll
  for (int h = 0; h < 8; ++h) {
    size_t idx = (h < 4 ? d : s) * 8 + h;
    float ex = __expf(eb[h] - Mx[idx]);
    eb[h] = ex;
    unsafeAtomicAdd(&Den[idx], ex);
  }
}

// ---------------------------------------------------------------- edge pass 3: alpha-weighted scatter (wave per edge)
__global__ __launch_bounds__(256) void edge_pass3(
    const long long* __restrict__ ei, const float* __restrict__ Ebuf,
    const float* __restrict__ Den, const __bf16* __restrict__ H,
    float* __restrict__ Agg, int E) {
  int wave = threadIdx.x >> 5;
  int lane = threadIdx.x & 31;
  int e = blockIdx.x * 8 + wave;
  if (e >= E) return;
  size_t s = (size_t)ei[e];
  size_t d = (size_t)ei[(size_t)E + e];
  const float* eb = Ebuf + (size_t)e * 8;
  float af[4], ab[4];
#pragma unroll
  for (int h = 0; h < 4; ++h) {
    af[h] = eb[h]     / (Den[d * 8 + h]     + 1e-16f);
    ab[h] = eb[4 + h] / (Den[s * 8 + 4 + h] + 1e-16f);
  }
  const __bf16* hs = H + s * NDIM;          // fwd message features (cols 0..255) of src
  const __bf16* hd = H + d * NDIM + 256;    // bwd message features (cols 256..511) of dst
  float* od = Agg + d * NDIM;
  float* os = Agg + s * NDIM + 256;
#pragma unroll
  for (int f = 0; f < 8; ++f) {
    int feat = f * 32 + lane;               // 0..255 ; head = (f*32+lane)>>6 == f>>1 (uniform)
    unsafeAtomicAdd(od + feat, (float)hs[feat] * af[f >> 1]);
    unsafeAtomicAdd(os + feat, (float)hd[feat] * ab[f >> 1]);
  }
}

// ---------------------------------------------------------------- agg + gat bias -> bf16
__global__ void agg_to_bf16(const float* __restrict__ Agg, const float* __restrict__ bias_f,
                            const float* __restrict__ bias_b, __bf16* __restrict__ Aggb, int N) {
  size_t i = (size_t)blockIdx.x * blockDim.x + threadIdx.x;
  if (i >= (size_t)N * NDIM) return;
  int c = (int)(i & 511);
  float b = (c < 256) ? bias_f[c] : bias_b[c - 256];
  Aggb[i] = (__bf16)(Agg[i] + b);
}

// ---------------------------------------------------------------- batchnorm stats (sum, sumsq per feature)
__global__ __launch_bounds__(256) void bn_stats(const float* __restrict__ F,
                                                float* __restrict__ Bn, int N) {
  int f0 = threadIdx.x;                      // handles features f0 and f0+256
  int r0 = blockIdx.x * 128;
  float s0 = 0.f, q0 = 0.f, s1 = 0.f, q1 = 0.f;
  for (int r = 0; r < 128; ++r) {
    int row = r0 + r;
    if (row < N) {
      float a = F[(size_t)row * NDIM + f0];
      float b = F[(size_t)row * NDIM + f0 + 256];
      s0 += a; q0 += a * a;
      s1 += b; q1 += b * b;
    }
  }
  unsafeAtomicAdd(&Bn[f0], s0);
  unsafeAtomicAdd(&Bn[512 + f0], q0);
  unsafeAtomicAdd(&Bn[f0 + 256], s1);
  unsafeAtomicAdd(&Bn[512 + f0 + 256], q1);
}

// ---------------------------------------------------------------- batchnorm apply + relu (in place on d_out)
__global__ void bn_apply(float* __restrict__ F, const float* __restrict__ Bn,
                         const float* __restrict__ gamma, const float* __restrict__ beta, int N) {
  size_t i = (size_t)blockIdx.x * blockDim.x + threadIdx.x;
  if (i >= (size_t)N * NDIM) return;
  int f = (int)(i & 511);
  float invN = 1.0f / (float)N;
  float mean = Bn[f] * invN;
  float var = Bn[512 + f] * invN - mean * mean;
  float v = (F[i] - mean) * rsqrtf(var + BN_EPS) * gamma[f] + beta[f];
  F[i] = v > 0.f ? v : 0.f;
}

// ---------------------------------------------------------------- launch
extern "C" void kernel_launch(void* const* d_in, const int* in_sizes, int n_in,
                              void* d_out, int out_size, void* d_ws, size_t ws_size,
                              hipStream_t stream) {
  const float*     x     = (const float*)d_in[0];
  const long long* ei    = (const long long*)d_in[1];
  const float*     Wf    = (const float*)d_in[2];
  const float*     asf   = (const float*)d_in[3];
  const float*     adf   = (const float*)d_in[4];
  const float*     biasf = (const float*)d_in[5];
  const float*     Wb    = (const float*)d_in[6];
  const float*     asb   = (const float*)d_in[7];
  const float*     adb   = (const float*)d_in[8];
  const float*     biasb = (const float*)d_in[9];
  const float*     Wfuse = (const float*)d_in[10];
  const float*     bfuse = (const float*)d_in[11];
  const float*     gamma = (const float*)d_in[12];
  const float*     beta  = (const float*)d_in[13];
  const int N = in_sizes[0] / NDIM;
  const int E = in_sizes[1] / 2;
  (void)n_in; (void)out_size; (void)ws_size;

  char* w = (char*)d_ws;
  auto take = [&](size_t bytes) { char* p = w; w += (bytes + 255) & ~(size_t)255; return p; };
  __bf16* xb   = (__bf16*)take((size_t)N * NDIM * 2);
  __bf16* WcT  = (__bf16*)take((size_t)NDIM * NDIM * 2);
  __bf16* WfT  = (__bf16*)take((size_t)NDIM * NDIM * 2);
  __bf16* Hb   = (__bf16*)take((size_t)N * NDIM * 2);
  float*  Asc  = (float*)take((size_t)N * 16 * 4);
  float*  Mx   = (float*)take((size_t)N * 8 * 4);
  float*  Den  = (float*)take((size_t)N * 8 * 4);
  float*  Ebuf = (float*)take((size_t)E * 8 * 4);
  float*  Agg  = (float*)take((size_t)N * NDIM * 4);
  __bf16* Aggb = (__bf16*)take((size_t)N * NDIM * 2);
  float*  Bn   = (float*)take(1024 * 4);
  float*  Out  = (float*)d_out;

  size_t tot = (size_t)N * NDIM;
  int blocksTot = (int)((tot + 255) / 256);
  dim3 gGemm((N + 127) / 128, NDIM / 128);

  prep_kernel<<<blocksTot, 256, 0, stream>>>(x, Wf, Wb, Wfuse, xb, WcT, WfT, N);
  init_ws_kernel<<<blocksTot, 256, 0, stream>>>(Agg, Mx, Den, Bn, N);
  wmma_gemm<true><<<gGemm, 256, 0, stream>>>(xb, WcT, nullptr, Hb, nullptr, N);
  node_scores<<<(N * 4 + 255) / 256, 256, 0, stream>>>(Hb, asf, adf, asb, adb, Asc, N);
  edge_pass1<<<(E + 255) / 256, 256, 0, stream>>>(ei, Asc, Ebuf, Mx, E);
  edge_pass2<<<(E + 255) / 256, 256, 0, stream>>>(ei, Mx, Ebuf, Den, E);
  edge_pass3<<<(E + 7) / 8, 256, 0, stream>>>(ei, Ebuf, Den, Hb, Agg, E);
  agg_to_bf16<<<blocksTot, 256, 0, stream>>>(Agg, biasf, biasb, Aggb, N);
  wmma_gemm<false><<<gGemm, 256, 0, stream>>>(Aggb, WfT, Out, nullptr, bfuse, N);
  bn_stats<<<(N + 127) / 128, 256, 0, stream>>>(Out, Bn, N);
  bn_apply<<<blocksTot, 256, 0, stream>>>(Out, Bn, gamma, beta, N);
}
